// LlamaAttention_30116310680230
// MI455X (gfx1250) — compile-verified, matrix operand order checked by reading, not run on hardware
//
#include <hip/hip_runtime.h>
#include <hip/hip_bf16.h>

// ---------------------------------------------------------------------------
// MI455X (gfx1250) fused attention block: QKV GEMM -> causal flash attention
// -> out GEMM + residual -> LayerNorm. All contractions via
// v_wmma_f32_16x16x32_f16 (wave32), fp32 accumulate, fp32 softmax/LN.
// All WMMA fragments are loaded as contiguous 16B LDS reads (ds_load_b128):
// B-operands are staged TRANSPOSED in LDS so no per-lane strided gathers.
// ---------------------------------------------------------------------------

typedef _Float16 half_t;
typedef half_t v16h __attribute__((ext_vector_type(16)));
typedef half_t v8h  __attribute__((ext_vector_type(8)));
typedef float  v8f  __attribute__((ext_vector_type(8)));

static __device__ __forceinline__ v8f wmma16x16x32(v16h a, v16h b, v8f c) {
    // (neg_a, A, neg_b, B, c_mod, C, reuse_a, reuse_b)
    return __builtin_amdgcn_wmma_f32_16x16x32_f16(false, a, false, b,
                                                  (short)0, c, false, false);
}

static __device__ __forceinline__ v16h cat8(v8h lo, v8h hi) {
    return __builtin_shufflevector(lo, hi, 0, 1, 2, 3, 4, 5, 6, 7,
                                           8, 9, 10, 11, 12, 13, 14, 15);
}

// A fragment, 16x32 (MxK), tile row-major, row stride `ld` (elements).
// ISA layout: lanes 0-15 row M=lane, K=0..7 & 16..23; lanes 16-31 row M=lane-16,
// K=8..15 & 24..31.  Requires: ld % 8 == 0 and p 16B-aligned -> 2x ds_load_b128.
static __device__ __forceinline__ v16h ld_frag_a(const half_t* p, int ld) {
    const int lane = threadIdx.x & 31;
    const int row  = lane & 15;
    const int kh   = (lane >> 4) << 3;   // 0 or 8
    const half_t* r = p + row * ld + kh;
    v8h lo = *(const v8h*)(r);
    v8h hi = *(const v8h*)(r + 16);
    return cat8(lo, hi);
}

// B fragment, 32x16 (KxN), read from a TRANSPOSED tile (tile[n][k], row
// stride `ld`).  Column N = lane&15; same K split as A -> contiguous reads.
static __device__ __forceinline__ v16h ld_frag_bT(const half_t* p, int ld) {
    const int lane = threadIdx.x & 31;
    const int col  = lane & 15;
    const int kh   = (lane >> 4) << 3;
    const half_t* r = p + col * ld + kh;
    v8h lo = *(const v8h*)(r);
    v8h hi = *(const v8h*)(r + 16);
    return cat8(lo, hi);
}

// ---------------------------------------------------------------------------
// Kernel 0: fp32 -> f16 conversion (grid-stride)
// ---------------------------------------------------------------------------
__global__ void cvt_f32_f16_kernel(const float* __restrict__ src,
                                   half_t* __restrict__ dst, int n) {
    int i = blockIdx.x * blockDim.x + threadIdx.x;
    int stride = gridDim.x * blockDim.x;
    for (; i < n; i += stride) dst[i] = (half_t)src[i];
}

// ---------------------------------------------------------------------------
// Kernel 1: QKV GEMM.  C[4096,3072] = X[4096,1024] @ W[1024,3072] + b.
// CTA tile 128x128, BK=32, 256 threads = 8 waves (4 M x 2 N), 32x64 per wave.
// B tile staged transposed ([n][k]) in LDS.  Epilogue scatters into q/k/v
// [B,H,T,Hd] f16; q pre-scaled by 1/sqrt(64).
// ---------------------------------------------------------------------------
__global__ __launch_bounds__(256) void qkv_gemm_kernel(
    const half_t* __restrict__ X, const half_t* __restrict__ W,
    const float* __restrict__ bias,
    half_t* __restrict__ qb, half_t* __restrict__ kbuf, half_t* __restrict__ vbuf) {
    constexpr int LDA = 40, LDBT = 40;
    __shared__ alignas(16) half_t lA[128 * LDA];     // [m][k]
    __shared__ alignas(16) half_t lBT[128 * LDBT];   // [n][k] (transposed)

    const int blockM = blockIdx.y * 128;
    const int blockN = blockIdx.x * 128;
    const int lane = threadIdx.x & 31;
    const int wave = threadIdx.x >> 5;
    const int wm = wave & 3;     // 4 waves along M (32 rows each)
    const int wn = wave >> 2;    // 2 waves along N (64 cols each)

    v8f acc[2][4] = {};

    for (int k0 = 0; k0 < 1024; k0 += 32) {
        {   // stage A: 128x32, 2 threads per row, 16 halfs each (b128 in/out)
            int row = threadIdx.x >> 1, hseg = (threadIdx.x & 1) * 16;
            const half_t* ap = X + (size_t)(blockM + row) * 1024 + k0 + hseg;
            half_t* la = lA + row * LDA + hseg;
            *(v8h*)(la)     = *(const v8h*)(ap);
            *(v8h*)(la + 8) = *(const v8h*)(ap + 8);
            if (k0 + 32 < 1024) __builtin_prefetch(ap + 32, 0, 0);
        }
        {   // stage B transposed: 32x128 global rows -> lBT[n][k]
            int krow = threadIdx.x >> 3, seg = (threadIdx.x & 7) * 16;
            const half_t* bp = W + (size_t)(k0 + krow) * 3072 + blockN + seg;
            v8h g0 = *(const v8h*)(bp);
            v8h g1 = *(const v8h*)(bp + 8);
#pragma unroll
            for (int i = 0; i < 8; ++i) {
                lBT[(seg + i) * LDBT + krow]     = g0[i];
                lBT[(seg + 8 + i) * LDBT + krow] = g1[i];
            }
            if (k0 + 32 < 1024) __builtin_prefetch(bp + 32 * 3072, 0, 0);
        }
        __syncthreads();

        v16h af[2];
#pragma unroll
        for (int mt = 0; mt < 2; ++mt)
            af[mt] = ld_frag_a(lA + (wm * 32 + mt * 16) * LDA, LDA);
#pragma unroll
        for (int nt = 0; nt < 4; ++nt) {
            v16h bf = ld_frag_bT(lBT + (wn * 64 + nt * 16) * LDBT, LDBT);
#pragma unroll
            for (int mt = 0; mt < 2; ++mt)
                acc[mt][nt] = wmma16x16x32(af[mt], bf, acc[mt][nt]);
        }
        __syncthreads();
    }

    // Epilogue: C-frag layout -> scatter into q/k/v [B*H, T, Hd]
    const int mb  = (lane >> 4) << 3;
    const int ncl = lane & 15;
#pragma unroll
    for (int mt = 0; mt < 2; ++mt)
#pragma unroll
        for (int nt = 0; nt < 4; ++nt)
#pragma unroll
            for (int r = 0; r < 8; ++r) {
                int m = blockM + wm * 32 + mt * 16 + mb + r;
                int n = blockN + wn * 64 + nt * 16 + ncl;
                float val = acc[mt][nt][r] + bias[n];
                int c = n >> 10, rem = n & 1023;
                int hh = rem >> 6, d = rem & 63;
                int bidx = m >> 11, t = m & 2047;
                size_t dst = (((size_t)(bidx * 16 + hh)) * 2048 + t) * 64 + d;
                half_t* p = (c == 0) ? qb : (c == 1 ? kbuf : vbuf);
                float sc = (c == 0) ? 0.125f : 1.0f;   // q pre-scaled by 1/sqrt(Hd)
                p[dst] = (half_t)(val * sc);
            }
}

// ---------------------------------------------------------------------------
// Kernel 2: causal flash attention.  Grid: (T/128, B*H).  128 queries/block,
// 8 waves, each wave owns a 16-row Q tile.  Key blocks of 64 staged in LDS:
// K as [key][d] (already the transposed layout for S = Q K^T fragments),
// V staged transposed as [d][key] so P@V fragments are contiguous too.
// ---------------------------------------------------------------------------
__global__ __launch_bounds__(256) void attn_kernel(
    const half_t* __restrict__ q, const half_t* __restrict__ k,
    const half_t* __restrict__ v, half_t* __restrict__ y) {
    constexpr int LDK = 72;
    __shared__ alignas(16) half_t lK[64 * LDK];         // [key][d]
    __shared__ alignas(16) half_t lVT[64 * LDK];        // [d][key]
    __shared__ alignas(16) half_t lP[8 * 16 * LDK];     // per-wave P (A-layout)

    const int bh = blockIdx.y;            // b*16 + h
    const int b  = bh >> 4, h = bh & 15;
    const int qbase = blockIdx.x * 128;
    const int lane = threadIdx.x & 31;
    const int wave = threadIdx.x >> 5;
    const int mb  = (lane >> 4) << 3;     // C-frag row base
    const int ncl = lane & 15;            // C-frag column

    const size_t headOff = (size_t)bh * 2048 * 64;

    // Load this wave's Q fragments (16 rows x 64, two K-steps) from global.
    const int qrow = qbase + wave * 16 + (lane & 15);
    const half_t* qp = q + headOff + (size_t)qrow * 64;
    const int kh = (lane >> 4) << 3;
    v16h qa[2];
#pragma unroll
    for (int s = 0; s < 2; ++s) {
        v8h lo = *(const v8h*)(qp + s * 32 + kh);
        v8h hi = *(const v8h*)(qp + s * 32 + 16 + kh);
        qa[s] = cat8(lo, hi);
    }

    v8f o[4] = {};
    float mrow[8], lrow[8];
#pragma unroll
    for (int r = 0; r < 8; ++r) { mrow[r] = -3.0e38f; lrow[r] = 0.0f; }

    const int nkb = (qbase + 128) >> 6;   // causal: only key blocks <= query tile

    for (int kb = 0; kb < nkb; ++kb) {
        {   // stage K [key][d] and V transposed [d][key]; 4 threads per row
            int row = threadIdx.x >> 2, seg = (threadIdx.x & 3) * 16;
            const half_t* kp = k + headOff + (size_t)(kb * 64 + row) * 64 + seg;
            const half_t* vp = v + headOff + (size_t)(kb * 64 + row) * 64 + seg;
            half_t* lk = lK + row * LDK + seg;
            *(v8h*)(lk)     = *(const v8h*)(kp);
            *(v8h*)(lk + 8) = *(const v8h*)(kp + 8);
            v8h g0 = *(const v8h*)(vp);
            v8h g1 = *(const v8h*)(vp + 8);
#pragma unroll
            for (int i = 0; i < 8; ++i) {
                lVT[(seg + i) * LDK + row]     = g0[i];
                lVT[(seg + 8 + i) * LDK + row] = g1[i];
            }
        }
        __syncthreads();

        // S = Q @ K^T  (contraction over Hd=64 -> two WMMA steps per key tile)
        v8f s[4] = {};
#pragma unroll
        for (int nt = 0; nt < 4; ++nt)
#pragma unroll
            for (int st = 0; st < 2; ++st) {
                v16h bf = ld_frag_bT(lK + (nt * 16) * LDK + st * 32, LDK);
                s[nt] = wmma16x16x32(qa[st], bf, s[nt]);
            }

        // Causal mask + online softmax (fp32, half-wave row reductions)
        float mnew[8];
#pragma unroll
        for (int r = 0; r < 8; ++r) mnew[r] = mrow[r];
#pragma unroll
        for (int nt = 0; nt < 4; ++nt)
#pragma unroll
            for (int r = 0; r < 8; ++r) {
                int kj = kb * 64 + nt * 16 + ncl;
                int qi = qbase + wave * 16 + mb + r;
                float val = s[nt][r];
                if (kj > qi) val = -3.0e38f;
                s[nt][r] = val;
                mnew[r] = fmaxf(mnew[r], val);
            }
#pragma unroll
        for (int r = 0; r < 8; ++r)
#pragma unroll
            for (int msk = 8; msk >= 1; msk >>= 1)
                mnew[r] = fmaxf(mnew[r], __shfl_xor(mnew[r], msk, 16));

        float alpha[8], rs[8];
#pragma unroll
        for (int r = 0; r < 8; ++r) { alpha[r] = __expf(mrow[r] - mnew[r]); rs[r] = 0.0f; }
#pragma unroll
        for (int nt = 0; nt < 4; ++nt)
#pragma unroll
            for (int r = 0; r < 8; ++r) {
                float p = __expf(s[nt][r] - mnew[r]);
                s[nt][r] = p;
                rs[r] += p;
            }
#pragma unroll
        for (int r = 0; r < 8; ++r)
#pragma unroll
            for (int msk = 8; msk >= 1; msk >>= 1)
                rs[r] += __shfl_xor(rs[r], msk, 16);
#pragma unroll
        for (int r = 0; r < 8; ++r) {
            lrow[r] = lrow[r] * alpha[r] + rs[r];
            mrow[r] = mnew[r];
        }
#pragma unroll
        for (int nt = 0; nt < 4; ++nt)
#pragma unroll
            for (int r = 0; r < 8; ++r) o[nt][r] *= alpha[r];

        // Re-shape P: C-frag -> LDS (row-major [m][key]) -> A-frag
        half_t* pw = lP + wave * 16 * LDK;
#pragma unroll
        for (int nt = 0; nt < 4; ++nt)
#pragma unroll
            for (int r = 0; r < 8; ++r)
                pw[(mb + r) * LDK + nt * 16 + ncl] = (half_t)s[nt][r];
        __syncthreads();

        // O += P @ V   (contraction over 64 keys -> two WMMA steps)
#pragma unroll
        for (int st = 0; st < 2; ++st) {
            v16h pa = ld_frag_a(pw + st * 32, LDK);
#pragma unroll
            for (int nt = 0; nt < 4; ++nt) {
                v16h bf = ld_frag_bT(lVT + (nt * 16) * LDK + st * 32, LDK);
                o[nt] = wmma16x16x32(pa, bf, o[nt]);
            }
        }
        __syncthreads();   // protect lK/lVT/lP before next iteration's staging
    }

    // Normalize and write y [B,T,D] f16 (column = h*64 + d)
#pragma unroll
    for (int nt = 0; nt < 4; ++nt)
#pragma unroll
        for (int r = 0; r < 8; ++r) {
            int t = qbase + wave * 16 + mb + r;
            float val = o[nt][r] * (1.0f / lrow[r]);
            y[((size_t)(b * 2048 + t)) * 1024 + h * 64 + nt * 16 + ncl] = (half_t)val;
        }
}

// ---------------------------------------------------------------------------
// Kernel 3: output GEMM + bias + residual.  h[4096,1024] = Y @ W_out + b + x.
// ---------------------------------------------------------------------------
__global__ __launch_bounds__(256) void out_gemm_kernel(
    const half_t* __restrict__ Y, const half_t* __restrict__ W,
    const float* __restrict__ bias, const float* __restrict__ x,
    float* __restrict__ hbuf) {
    constexpr int LDA = 40, LDBT = 40;
    __shared__ alignas(16) half_t lA[128 * LDA];
    __shared__ alignas(16) half_t lBT[128 * LDBT];

    const int blockM = blockIdx.y * 128;
    const int blockN = blockIdx.x * 128;
    const int lane = threadIdx.x & 31;
    const int wave = threadIdx.x >> 5;
    const int wm = wave & 3;
    const int wn = wave >> 2;

    v8f acc[2][4] = {};

    for (int k0 = 0; k0 < 1024; k0 += 32) {
        {
            int row = threadIdx.x >> 1, hseg = (threadIdx.x & 1) * 16;
            const half_t* ap = Y + (size_t)(blockM + row) * 1024 + k0 + hseg;
            half_t* la = lA + row * LDA + hseg;
            *(v8h*)(la)     = *(const v8h*)(ap);
            *(v8h*)(la + 8) = *(const v8h*)(ap + 8);
            if (k0 + 32 < 1024) __builtin_prefetch(ap + 32, 0, 0);
        }
        {
            int krow = threadIdx.x >> 3, seg = (threadIdx.x & 7) * 16;
            const half_t* bp = W + (size_t)(k0 + krow) * 1024 + blockN + seg;
            v8h g0 = *(const v8h*)(bp);
            v8h g1 = *(const v8h*)(bp + 8);
#pragma unroll
            for (int i = 0; i < 8; ++i) {
                lBT[(seg + i) * LDBT + krow]     = g0[i];
                lBT[(seg + 8 + i) * LDBT + krow] = g1[i];
            }
            if (k0 + 32 < 1024) __builtin_prefetch(bp + 32 * 1024, 0, 0);
        }
        __syncthreads();

        v16h af[2];
#pragma unroll
        for (int mt = 0; mt < 2; ++mt)
            af[mt] = ld_frag_a(lA + (wm * 32 + mt * 16) * LDA, LDA);
#pragma unroll
        for (int nt = 0; nt < 4; ++nt) {
            v16h bf = ld_frag_bT(lBT + (wn * 64 + nt * 16) * LDBT, LDBT);
#pragma unroll
            for (int mt = 0; mt < 2; ++mt)
                acc[mt][nt] = wmma16x16x32(af[mt], bf, acc[mt][nt]);
        }
        __syncthreads();
    }

    const int mb  = (lane >> 4) << 3;
    const int ncl = lane & 15;
#pragma unroll
    for (int mt = 0; mt < 2; ++mt)
#pragma unroll
        for (int nt = 0; nt < 4; ++nt)
#pragma unroll
            for (int r = 0; r < 8; ++r) {
                int m = blockM + wm * 32 + mt * 16 + mb + r;
                int n = blockN + wn * 64 + nt * 16 + ncl;
                size_t idx = (size_t)m * 1024 + n;
                hbuf[idx] = acc[mt][nt][r] + bias[n] + x[idx];
            }
}

// ---------------------------------------------------------------------------
// Kernel 4: LayerNorm, one block per token row (1024 elements).
// ---------------------------------------------------------------------------
__global__ __launch_bounds__(256) void ln_kernel(
    const float* __restrict__ hbuf, const float* __restrict__ gamma,
    const float* __restrict__ beta, float* __restrict__ out) {
    const int row = blockIdx.x;
    const float* hr = hbuf + (size_t)row * 1024;

    float s = 0.0f, s2 = 0.0f;
    for (int i = threadIdx.x; i < 1024; i += 256) {
        float v = hr[i];
        s += v; s2 += v * v;
    }
#pragma unroll
    for (int msk = 16; msk >= 1; msk >>= 1) {
        s  += __shfl_xor(s,  msk, 32);
        s2 += __shfl_xor(s2, msk, 32);
    }
    __shared__ float red[2][8];
    if ((threadIdx.x & 31) == 0) {
        red[0][threadIdx.x >> 5] = s;
        red[1][threadIdx.x >> 5] = s2;
    }
    __syncthreads();
    s = 0.0f; s2 = 0.0f;
#pragma unroll
    for (int w = 0; w < 8; ++w) { s += red[0][w]; s2 += red[1][w]; }

    const float mu  = s * (1.0f / 1024.0f);
    const float var = s2 * (1.0f / 1024.0f) - mu * mu;
    const float inv = rsqrtf(var + 1e-5f);
    for (int i = threadIdx.x; i < 1024; i += 256)
        out[(size_t)row * 1024 + i] = (hr[i] - mu) * inv * gamma[i] + beta[i];
}

// ---------------------------------------------------------------------------
// Host-side launcher
// ---------------------------------------------------------------------------
extern "C" void kernel_launch(void* const* d_in, const int* in_sizes, int n_in,
                              void* d_out, int out_size, void* d_ws, size_t ws_size,
                              hipStream_t stream) {
    (void)in_sizes; (void)n_in; (void)out_size; (void)ws_size;

    const float* x     = (const float*)d_in[0];   // [2,2048,1024]
    const float* W_qkv = (const float*)d_in[1];   // [1024,3072]
    const float* b_qkv = (const float*)d_in[2];   // [3072]
    const float* W_out = (const float*)d_in[3];   // [1024,1024]
    const float* b_out = (const float*)d_in[4];   // [1024]
    const float* gamma = (const float*)d_in[5];   // [1024]
    const float* beta  = (const float*)d_in[6];   // [1024]
    float* out = (float*)d_out;                   // [2,2048,1024]

    // Workspace carve-up (all offsets 1MB-aligned)
    char* ws = (char*)d_ws;
    half_t* xb    = (half_t*)(ws);                         // 4096*1024   f16 (8 MB)
    half_t* wqkvb = (half_t*)(ws + (8u << 20));            // 1024*3072   f16 (6 MB)
    half_t* woutb = (half_t*)(ws + (14u << 20));           // 1024*1024   f16 (2 MB)
    half_t* qb    = (half_t*)(ws + (16u << 20));           // [B,H,T,Hd]  f16 (8 MB)
    half_t* kb    = (half_t*)(ws + (24u << 20));           // 8 MB
    half_t* vb    = (half_t*)(ws + (32u << 20));           // 8 MB
    half_t* yb    = (half_t*)(ws + (40u << 20));           // [B,T,D] f16 (8 MB)
    float*  hbuf  = (float*)(ws + (48u << 20));            // [B,T,D] f32 (16 MB)

    // 0) fp32 -> f16 conversions
    cvt_f32_f16_kernel<<<512, 256, 0, stream>>>(x,     xb,    4096 * 1024);
    cvt_f32_f16_kernel<<<512, 256, 0, stream>>>(W_qkv, wqkvb, 1024 * 3072);
    cvt_f32_f16_kernel<<<256, 256, 0, stream>>>(W_out, woutb, 1024 * 1024);

    // 1) QKV projection (M=4096, N=3072, K=1024)
    qkv_gemm_kernel<<<dim3(24, 32), 256, 0, stream>>>(xb, wqkvb, b_qkv, qb, kb, vb);

    // 2) causal flash attention (grid: qblocks x B*H)
    attn_kernel<<<dim3(16, 32), 256, 0, stream>>>(qb, kb, vb, yb);

    // 3) output projection + bias + residual (M=4096, N=1024, K=1024)
    out_gemm_kernel<<<dim3(8, 32), 256, 0, stream>>>(yb, woutb, b_out, x, hbuf);

    // 4) LayerNorm per token row
    ln_kernel<<<4096, 256, 0, stream>>>(hbuf, gamma, beta, out);
}
